// MoleculeEncoder_GCN_82016695485245
// MI455X (gfx1250) — compile-verified, matrix operand order checked by reading, not run on hardware
//
#include <hip/hip_runtime.h>
#include <hip/hip_bf16.h>
#include <math.h>

#define N_NODES  100000
#define N_EDGES  1600000
#define N_GRAPHS 2048
#define IN_C     54
#define HID      32
#define OUT_C    64
#define K1PAD    64          // IN_C padded to two bf16 WMMA K-steps

typedef __attribute__((ext_vector_type(16))) __bf16 v16bf;
typedef __attribute__((ext_vector_type(8)))  float  v8f;

__device__ __forceinline__ v16bf make_frag(uint4 lo, uint4 hi) {
    union { uint4 u[2]; v16bf v; } t;
    t.u[0] = lo; t.u[1] = hi;
    return t.v;
}

__device__ __forceinline__ unsigned pack_bf16_pair(float a, float b) {
    union { __bf16 h[2]; unsigned u; } t;
    t.h[0] = (__bf16)a; t.h[1] = (__bf16)b;
    return t.u;
}

__device__ __forceinline__ void atomic_add_f32(float* p, float v) {
    (void)__hip_atomic_fetch_add(p, v, __ATOMIC_RELAXED, __HIP_MEMORY_SCOPE_AGENT);
}

// ---------------------------------------------------------------- utilities
__global__ void fill_kernel(float* __restrict__ p, long n, float val) {
    long i = (long)blockIdx.x * blockDim.x + threadIdx.x;
    const long stride = (long)gridDim.x * blockDim.x;
    for (; i < n; i += stride) p[i] = val;
}

// pack x [N][54] f32 -> xpad [N][64] bf16 (zero padded K), pair-packed stores
__global__ void pack_x_kernel(const float* __restrict__ x, unsigned* __restrict__ xpad) {
    const long t = (long)blockIdx.x * blockDim.x + threadIdx.x;   // N * 32 pairs
    if (t >= (long)N_NODES * (K1PAD / 2)) return;
    const int row = (int)(t >> 5);
    const int k0  = (int)(t & 31) * 2;
    const float v0 = (k0     < IN_C) ? x[row * IN_C + k0]     : 0.0f;
    const float v1 = (k0 + 1 < IN_C) ? x[row * IN_C + k0 + 1] : 0.0f;
    xpad[t] = pack_bf16_pair(v0, v1);
}

// pack W1 [54][32] -> W1t [32][64] bf16 (transposed, K padded);
// pack W2 [32][64] -> W2t [64][32] bf16 (transposed). Single block.
__global__ void pack_w_kernel(const float* __restrict__ W1, const float* __restrict__ W2,
                              __bf16* __restrict__ w1t, __bf16* __restrict__ w2t) {
    for (int t = threadIdx.x; t < HID * K1PAD; t += blockDim.x) {
        const int n = t / K1PAD, k = t % K1PAD;
        w1t[t] = (k < IN_C) ? (__bf16)W1[k * HID + n] : (__bf16)0.0f;
    }
    for (int t = threadIdx.x; t < OUT_C * HID; t += blockDim.x) {
        const int n = t / HID, k = t % HID;
        w2t[t] = (__bf16)W2[k * OUT_C + n];
    }
}

// ------------------------------------------------------------ degree / dinv
__global__ void deg_kernel(const int* __restrict__ dst, float* __restrict__ deg) {
    const int e = blockIdx.x * blockDim.x + threadIdx.x;
    if (e >= N_EDGES) return;
    atomic_add_f32(&deg[dst[e]], 1.0f);
}

__global__ void rsqrt_kernel(float* __restrict__ deg) {
    const int i = blockIdx.x * blockDim.x + threadIdx.x;
    if (i >= N_NODES) return;
    deg[i] = rsqrtf(deg[i]);     // deg pre-filled with 1.0 (self-loop)
}

// ------------------------------------------------------- GEMM1: xw1 = x @ W1
// A/B layout (ISA 7.12.2): lane half hi selects K offset 8; elements 0..7 hold
// K = koff..koff+7, elements 8..15 hold K = 16+koff..23+koff -> two b128 loads.
__global__ void gemm1_wmma(const __bf16* __restrict__ xp,
                           const __bf16* __restrict__ w1t,
                           float* __restrict__ xw1) {
    const int lane    = threadIdx.x & 31;
    const int rowTile = blockIdx.x * (blockDim.x >> 5) + (threadIdx.x >> 5);
    if (rowTile >= N_NODES / 16) return;        // wave-uniform; EXEC stays full
    const int hi   = lane >> 4;
    const int l16  = lane & 15;
    const int row  = rowTile * 16 + l16;
    const int koff = hi * 8;

    const uint4* ap = (const uint4*)(xp + (size_t)row * K1PAD + koff);
    const v16bf a0 = make_frag(ap[0], ap[2]);   // K koff..+7, 16+koff..+7
    const v16bf a1 = make_frag(ap[4], ap[6]);   // K 32+koff.., 48+koff..

    #pragma unroll
    for (int nt = 0; nt < 2; ++nt) {
        const int n = nt * 16 + l16;
        const uint4* bp = (const uint4*)(w1t + (size_t)n * K1PAD + koff);
        const v16bf b0 = make_frag(bp[0], bp[2]);
        const v16bf b1 = make_frag(bp[4], bp[6]);
        v8f c = {};
        c = __builtin_amdgcn_wmma_f32_16x16x32_bf16(false, a0, false, b0,
                                                    (short)0, c, false, false);
        c = __builtin_amdgcn_wmma_f32_16x16x32_bf16(false, a1, false, b1,
                                                    (short)0, c, false, false);
        #pragma unroll
        for (int r = 0; r < 8; ++r) {
            const int m = rowTile * 16 + hi * 8 + r;
            xw1[m * HID + nt * 16 + l16] = c[r];
        }
    }
}

// ------------------------------------------------------ GEMM2: xw2 = h1 @ W2
__global__ void gemm2_wmma(const __bf16* __restrict__ h1b,
                           const __bf16* __restrict__ w2t,
                           float* __restrict__ xw2) {
    const int lane    = threadIdx.x & 31;
    const int rowTile = blockIdx.x * (blockDim.x >> 5) + (threadIdx.x >> 5);
    if (rowTile >= N_NODES / 16) return;
    const int hi   = lane >> 4;
    const int l16  = lane & 15;
    const int row  = rowTile * 16 + l16;
    const int koff = hi * 8;

    const uint4* ap = (const uint4*)(h1b + (size_t)row * HID + koff);
    const v16bf a = make_frag(ap[0], ap[2]);    // exact K = 0..31 coverage

    #pragma unroll
    for (int nt = 0; nt < 4; ++nt) {
        const int n = nt * 16 + l16;
        const uint4* bp = (const uint4*)(w2t + (size_t)n * HID + koff);
        const v16bf b = make_frag(bp[0], bp[2]);
        v8f c = {};
        c = __builtin_amdgcn_wmma_f32_16x16x32_bf16(false, a, false, b,
                                                    (short)0, c, false, false);
        #pragma unroll
        for (int r = 0; r < 8; ++r) {
            const int m = rowTile * 16 + hi * 8 + r;
            xw2[m * OUT_C + nt * 16 + l16] = c[r];
        }
    }
}

// ----------------------------------------------- edge scatter: agg += w*xw[src]
template <int C>
__global__ void edge_agg_kernel(const int* __restrict__ src,
                                const int* __restrict__ dst,
                                const float* __restrict__ dinv,
                                const float* __restrict__ xw,
                                float* __restrict__ agg) {
    constexpr int GP = C / 4;
    const long tid = (long)blockIdx.x * blockDim.x + threadIdx.x;
    if (tid >= (long)N_EDGES * GP) return;
    const int e = (int)(tid / GP);
    const int g = (int)(tid % GP);
    const int s = src[e];
    const int d = dst[e];
    const float w = dinv[s] * dinv[d];
    const float4 v = *(const float4*)(xw + (long)s * C + g * 4);
    float* p = agg + (long)d * C + g * 4;
    atomic_add_f32(p + 0, w * v.x);
    atomic_add_f32(p + 1, w * v.y);
    atomic_add_f32(p + 2, w * v.z);
    atomic_add_f32(p + 3, w * v.w);
}

// ---- epilogue 1: h1 = relu(agg1 + dinv^2*xw1 + b1), written as packed bf16 pairs
__global__ void epilogue1_kernel(const float* __restrict__ xw1,
                                 const float* __restrict__ agg1,
                                 const float* __restrict__ dinv,
                                 const float* __restrict__ b1,
                                 unsigned* __restrict__ h1b) {
    const long t = (long)blockIdx.x * blockDim.x + threadIdx.x;   // N * 16 pairs
    if (t >= (long)N_NODES * (HID / 2)) return;
    const int i  = (int)(t >> 4);
    const int c0 = (int)(t & 15) * 2;
    const float di = dinv[i];
    const float d2 = di * di;
    const long base = (long)i * HID + c0;
    const float v0 = fmaxf(agg1[base]     + d2 * xw1[base]     + b1[c0],     0.0f);
    const float v1 = fmaxf(agg1[base + 1] + d2 * xw1[base + 1] + b1[c0 + 1], 0.0f);
    h1b[t] = pack_bf16_pair(v0, v1);
}

// -------- epilogue 2: h2 = agg2 + dinv^2*xw2 + b2; mish; pool segment-sum
__global__ void epilogue2_pool_kernel(const float* __restrict__ xw2,
                                      const float* __restrict__ agg2,
                                      const float* __restrict__ dinv,
                                      const float* __restrict__ b2,
                                      const int* __restrict__ batch,
                                      float* __restrict__ psum) {
    const long t = (long)blockIdx.x * blockDim.x + threadIdx.x;
    if (t >= (long)N_NODES * OUT_C) return;
    const int i = (int)(t / OUT_C);
    const int c = (int)(t % OUT_C);
    const float di = dinv[i];
    const float v = agg2[t] + di * di * xw2[t] + b2[c];
    const float sp = log1pf(expf(v));            // softplus (safe saturation)
    const float m  = v * tanhf(sp);              // mish
    atomic_add_f32(&psum[(long)batch[i] * OUT_C + c], m);
}

__global__ void count_kernel(const int* __restrict__ batch, float* __restrict__ pcnt) {
    const int i = blockIdx.x * blockDim.x + threadIdx.x;
    if (i >= N_NODES) return;
    atomic_add_f32(&pcnt[batch[i]], 1.0f);
}

__global__ void final_kernel(const float* __restrict__ psum,
                             const float* __restrict__ pcnt,
                             float* __restrict__ out) {
    const int t = blockIdx.x * blockDim.x + threadIdx.x;
    if (t >= N_GRAPHS * OUT_C) return;
    out[t] = psum[t] / fmaxf(pcnt[t / OUT_C], 1.0f);
}

// --------------------------------------------------------------------- launch
extern "C" void kernel_launch(void* const* d_in, const int* in_sizes, int n_in,
                              void* d_out, int out_size, void* d_ws, size_t ws_size,
                              hipStream_t stream) {
    const float* x   = (const float*)d_in[0];
    const int*   ei  = (const int*)d_in[1];      // [2, E] row-major
    const int*   src = ei;
    const int*   dst = ei + N_EDGES;
    const int*   bat = (const int*)d_in[2];
    const float* W1  = (const float*)d_in[3];
    const float* b1  = (const float*)d_in[4];
    const float* W2  = (const float*)d_in[5];
    const float* b2  = (const float*)d_in[6];
    float* out = (float*)d_out;

    // ---- workspace layout (float units; all regions 16B aligned) ----
    float* cur  = (float*)d_ws;
    float* dinv = cur;  cur += N_NODES;                       // N
    float* bufA = cur;  cur += (size_t)N_NODES * 64;          // xpad(32N) | xw1(32N); agg2 later
    __bf16* xpad = (__bf16*)bufA;                             //   N*64 bf16
    float*  xw1  = bufA + (size_t)N_NODES * 32;               //   32N f32
    float*  agg2 = bufA;                                      //   64N f32 (aliases, post-GEMM2)
    float* agg1 = cur;  cur += (size_t)N_NODES * HID;         // 32N
    __bf16* h1b = (__bf16*)cur; cur += (size_t)N_NODES * (HID / 2);  // N*32 bf16
    float* xw2  = cur;  cur += (size_t)N_NODES * OUT_C;       // 64N
    __bf16* w1t = (__bf16*)cur; cur += (HID * K1PAD) / 2;     // 2048 bf16
    __bf16* w2t = (__bf16*)cur; cur += (OUT_C * HID) / 2;     // 2048 bf16
    float* psum = cur;  cur += (size_t)N_GRAPHS * OUT_C;
    float* pcnt = cur;  cur += N_GRAPHS;

    const int B = 256;
    const int zblk = 4096;

    // init accumulators (every call; no state carried across launches)
    fill_kernel<<<zblk, B, 0, stream>>>(dinv, (long)N_NODES, 1.0f);
    fill_kernel<<<zblk, B, 0, stream>>>(agg1, (long)N_NODES * HID, 0.0f);
    fill_kernel<<<zblk, B, 0, stream>>>(psum, (long)N_GRAPHS * OUT_C, 0.0f);
    fill_kernel<<<zblk, B, 0, stream>>>(pcnt, (long)N_GRAPHS, 0.0f);

    // pack operands to WMMA-friendly bf16 layouts
    {
        const long work = (long)N_NODES * (K1PAD / 2);
        pack_x_kernel<<<(int)((work + B - 1) / B), B, 0, stream>>>(x, (unsigned*)xpad);
    }
    pack_w_kernel<<<1, B, 0, stream>>>(W1, W2, w1t, w2t);

    // degree + rsqrt
    deg_kernel<<<(N_EDGES + B - 1) / B, B, 0, stream>>>(dst, dinv);
    rsqrt_kernel<<<(N_NODES + B - 1) / B, B, 0, stream>>>(dinv);

    // layer 1
    const int rowTiles = N_NODES / 16;               // 6250 exactly
    const int gemmGrid = (rowTiles + (B / 32) - 1) / (B / 32);
    gemm1_wmma<<<gemmGrid, B, 0, stream>>>(xpad, w1t, xw1);
    {
        const long work = (long)N_EDGES * (HID / 4);
        edge_agg_kernel<HID><<<(int)((work + B - 1) / B), B, 0, stream>>>(
            src, dst, dinv, xw1, agg1);
    }
    {
        const long work = (long)N_NODES * (HID / 2);
        epilogue1_kernel<<<(int)((work + B - 1) / B), B, 0, stream>>>(
            xw1, agg1, dinv, b1, (unsigned*)h1b);
    }

    // layer 2
    gemm2_wmma<<<gemmGrid, B, 0, stream>>>(h1b, w2t, xw2);
    fill_kernel<<<zblk, B, 0, stream>>>(agg2, (long)N_NODES * OUT_C, 0.0f);
    {
        const long work = (long)N_EDGES * (OUT_C / 4);
        edge_agg_kernel<OUT_C><<<(int)((work + B - 1) / B), B, 0, stream>>>(
            src, dst, dinv, xw2, agg2);
    }

    // mish + global mean pool
    epilogue2_pool_kernel<<<(int)(((long)N_NODES * OUT_C + B - 1) / B), B, 0, stream>>>(
        xw2, agg2, dinv, b2, bat, psum);
    count_kernel<<<(N_NODES + B - 1) / B, B, 0, stream>>>(bat, pcnt);
    final_kernel<<<(N_GRAPHS * OUT_C + B - 1) / B, B, 0, stream>>>(psum, pcnt, out);
}
